// TinyCausalWanModel_81741817577508
// MI455X (gfx1250) — matrix operand order repaired and checked by reading, not compiled, not run on hardware
//
#include <hip/hip_runtime.h>

// ---------------------------------------------------------------- constants
static constexpr int S   = 1560;   // 3*20*26 tokens
static constexpr int DIM = 1536;
static constexpr int NH  = 12;
static constexpr int HD  = 128;
static constexpr int HW  = 520;    // H*W (frame size)
static constexpr int Wd  = 26;
static constexpr int Hd  = 20;
static constexpr int C2  = 64;     // HD/2 rope pairs; freqs row length
// rope split: CT=22 (time), CH=21 (height), CW=21 (width)

typedef __attribute__((ext_vector_type(16))) __bf16 v16bf;
typedef __attribute__((ext_vector_type(8)))  float  v8f;
typedef __attribute__((ext_vector_type(4)))  unsigned int u32x4;
typedef __attribute__((ext_vector_type(8)))  int          i32x8;
typedef __attribute__((ext_vector_type(4)))  int          i32x4;

__device__ inline v8f v8f_zero() {
  v8f z = {0.f,0.f,0.f,0.f,0.f,0.f,0.f,0.f};
  return z;
}

__device__ inline v8f wmma_bf16(v16bf a, v16bf b, v8f c) {
  // (neg_a, A, neg_b, B, c_mod, C, reuse_a, reuse_b)
  return __builtin_amdgcn_wmma_f32_16x16x32_bf16(false, a, false, b, (short)0, c, false, false);
}

// ------------------------------------------------------------ TDM 2D tile DMA
// Loads a tile of 16-bit elements (tile_d0 x tile_d1) from a row-major 2D
// tensor (remaining extent tensor_d0 x tensor_d1, row stride stride0 elems)
// into LDS at lds_addr. OOB rows/cols are zero-filled by the TDM engine.
// If pad80 is set, inserts 16B of LDS padding after each 64B of data
// (row pitch 80B == __bf16[..][40] tiles). Descriptor layout per ISA §8.3/8.4.
__device__ inline void tdm_load_2d_b16(unsigned lds_addr, const void* gptr,
                                       unsigned tensor_d0, unsigned tensor_d1,
                                       unsigned stride0,
                                       unsigned tile_d0, unsigned tile_d1,
                                       bool pad80) {
  unsigned long long ga = (unsigned long long)gptr;
  u32x4 g0;
  g0[0] = 1u;                                   // count=1, user descriptor
  g0[1] = lds_addr;                             // lds_addr [63:32]
  g0[2] = (unsigned)(ga & 0xffffffffu);         // global_addr [95:64]
  g0[3] = (unsigned)((ga >> 32) & 0x01ffffffu)  // global_addr [120:96]
        | (2u << 30);                           // type = 2 ("image")

  unsigned d[8];
  d[0] = (1u << 16);                            // data_size = 2 bytes
  if (pad80) d[0] |= (1u << 20)                 // pad_enable
                   | (3u << 22)                 // pad_interval: 16 DWORDs (64B)
                   | (3u << 25);                // pad_amount:   4 DWORDs (16B)
  d[1] = (tensor_d0 & 0xffffu) << 16;           // tensor_dim0[15:0] @bits[63:48]
  d[2] = (tensor_d0 >> 16) | ((tensor_d1 & 0xffffu) << 16);
  d[3] = (tensor_d1 >> 16) | (tile_d0 << 16);   // tile_dim0 @bits[127:112]
  d[4] = tile_d1 & 0xffffu;                     // tile_dim1; tile_dim2 = 0
  d[5] = stride0;                               // tensor_dim0_stride[31:0]
  d[6] = 0u;                                    // stride[47:32], dim1_stride lo
  d[7] = 0u;

  i32x8 g1;
#pragma unroll
  for (int i = 0; i < 8; ++i) g1[i] = (int)d[i];
  i32x4 z4 = {0, 0, 0, 0};
#if defined(__clang_major__) && (__clang_major__ >= 23)
  i32x8 z8 = {0, 0, 0, 0, 0, 0, 0, 0};
  __builtin_amdgcn_tensor_load_to_lds(g0, g1, z4, z4, z8, 0);
#else
  __builtin_amdgcn_tensor_load_to_lds(g0, g1, z4, z4, 0);
#endif
}

__device__ inline unsigned lds_addr_of(const void* p) {
  // Flat LDS-aperture addresses carry the LDS byte offset in the low 32 bits.
  return (unsigned)(unsigned long long)(uintptr_t)p;
}

// A fragment (16x32 bf16, MxK). LDS source is row-major [16+ rows][ld].
// Lane L: M = L&15, half = L>>4.
// VGPR v: K pair base = (v>=4 ? 16:0) + half*8 + (v&3)*2   (ISA 7.12.2)
__device__ inline v16bf load_frag_a(const __bf16* p, int ld) {
  const int lane = threadIdx.x & 31;
  const __bf16* src = p + (lane & 15) * ld;
  const int half8 = (lane >> 4) << 3;
  v16bf a;
#pragma unroll
  for (int v = 0; v < 8; ++v) {
    int kb = ((v & 4) << 2) + half8 + ((v & 3) << 1);
    a[2*v]   = src[kb];
    a[2*v+1] = src[kb+1];
  }
  return a;
}

// B fragment (32x16 bf16, KxN), LDS source stored column-major-in-N:
// row n holds K contiguous (i.e. [n][k], row pitch ld).
// Lane L: N = L&15, elem i: K = (i>=8 ? 16:0) + (L>>4)*8 + (i&7)   (matches A).
__device__ inline v16bf load_frag_b(const __bf16* p, int ld) {
  const int lane = threadIdx.x & 31;
  const __bf16* src = p + (lane & 15) * ld + ((lane >> 4) << 3);
  v16bf b;
#pragma unroll
  for (int i = 0; i < 16; ++i) b[i] = src[((i & 8) << 1) + (i & 7)];
  return b;
}

// ---------------------------------------------------------------- convert
__global__ __launch_bounds__(256) void f32_to_bf16_kernel(
    const float* __restrict__ in, __bf16* __restrict__ out, int n) {
  for (int i = blockIdx.x * 256 + threadIdx.x; i < n; i += gridDim.x * 256)
    out[i] = (__bf16)in[i];
}

// ---------------------------------------------------------------- GEMM
// out[m][n] = sum_k A[m][k] * Wt[n][k] + bias[n]
// A: MxK bf16 row-major, Wt: NxK bf16 row-major (weight[e][d] as in einsum).
// Block 128 threads (4 waves), tile 64x64, K-step 32, each wave owns 32x32.
// Double-buffered TDM pipeline: wave 0 DMAs tile t+1 into the spare LDS
// buffer while all waves run WMMA on tile t; s_wait_tensorcnt 2 requires
// only the older (in-order) tile pair to be complete.
__global__ __launch_bounds__(128) void gemm_bf16_wmma(
    const __bf16* __restrict__ A, const __bf16* __restrict__ Wt,
    const float* __restrict__ bias, float* __restrict__ out,
    int M, int N, int K) {
  __shared__ __bf16 As[2][64][40];   // 80B rows (64B data + 16B TDM pad)
  __shared__ __bf16 Bs[2][64][40];

  const int tid  = threadIdx.x;
  const int lane = tid & 31;
  const int wave = tid >> 5;
  const int wm = wave & 1, wn = wave >> 1;
  const int m0 = blockIdx.x * 64;
  const int n0 = blockIdx.y * 64;
  const int nT = K >> 5;

  auto issue = [&](int t, int p) {
    const int k0 = t << 5;
    tdm_load_2d_b16(lds_addr_of(&As[p][0][0]), A + (size_t)m0 * K + k0,
                    (unsigned)(K - k0), (unsigned)(M - m0), (unsigned)K,
                    32u, 64u, true);
    tdm_load_2d_b16(lds_addr_of(&Bs[p][0][0]), Wt + (size_t)n0 * K + k0,
                    (unsigned)(K - k0), (unsigned)(N - n0), (unsigned)K,
                    32u, 64u, true);
  };

  v8f acc[2][2];
#pragma unroll
  for (int i = 0; i < 2; ++i)
#pragma unroll
    for (int j = 0; j < 2; ++j) acc[i][j] = v8f_zero();

  if (wave == 0) issue(0, 0);

  for (int t = 0; t < nT; ++t) {
    const int p = t & 1;
    if (wave == 0) {
      if (t + 1 < nT) {
        issue(t + 1, p ^ 1);                       // prefetch next tile pair
        __builtin_amdgcn_s_wait_tensorcnt((short)2);  // tile t pair complete
      } else {
        __builtin_amdgcn_s_wait_tensorcnt((short)0);
      }
    }
    __syncthreads();

    v16bf a0 = load_frag_a(&As[p][wm * 32 +  0][0], 40);
    v16bf a1 = load_frag_a(&As[p][wm * 32 + 16][0], 40);
    v16bf b0 = load_frag_b(&Bs[p][wn * 32 +  0][0], 40);
    v16bf b1 = load_frag_b(&Bs[p][wn * 32 + 16][0], 40);
    acc[0][0] = wmma_bf16(a0, b0, acc[0][0]);
    acc[0][1] = wmma_bf16(a0, b1, acc[0][1]);
    acc[1][0] = wmma_bf16(a1, b0, acc[1][0]);
    acc[1][1] = wmma_bf16(a1, b1, acc[1][1]);
    __syncthreads();   // all reads of buffer p done before it is re-filled
  }

#pragma unroll
  for (int mi = 0; mi < 2; ++mi)
#pragma unroll
    for (int ni = 0; ni < 2; ++ni) {
      int gn = n0 + wn * 32 + ni * 16 + (lane & 15);
      float bv = bias[gn];
#pragma unroll
      for (int v = 0; v < 8; ++v) {
        int gm = m0 + wm * 32 + mi * 16 + v + ((lane >> 4) << 3);
        if (gm < M) out[(size_t)gm * N + gn] = acc[mi][ni][v] + bv;
      }
    }
}

// ---------------------------------------------------------------- RMSnorm+RoPE
// One block per token row. Full-row (1536) RMS, then interleaved-pair rope.
__global__ __launch_bounds__(256) void rmsnorm_rope_kernel(
    const float* __restrict__ in, const float* __restrict__ nw,
    const float* __restrict__ freqs, __bf16* __restrict__ outb) {
  __shared__ float red[256];
  __shared__ float srs;
  const int s = blockIdx.x;
  const float* row = in + (size_t)s * DIM;

  float ss = 0.f;
  for (int i = threadIdx.x; i < DIM; i += 256) { float t = row[i]; ss += t * t; }
  red[threadIdx.x] = ss;
  __syncthreads();
  for (int st = 128; st > 0; st >>= 1) {
    if (threadIdx.x < st) red[threadIdx.x] += red[threadIdx.x + st];
    __syncthreads();
  }
  if (threadIdx.x == 0) srs = rsqrtf(red[0] / (float)DIM + 1e-6f);
  __syncthreads();
  const float rs = srs;

  const int t_idx = s / HW;
  const int h_idx = (s / Wd) % Hd;
  const int w_idx = s % Wd;

  for (int p = threadIdx.x; p < DIM / 2; p += 256) {
    int head = p >> 6;          // 64 pairs per head
    int c    = p & 63;          // pair index within head
    int base = head * HD + c * 2;
    float x0 = row[base]     * rs * nw[base];
    float x1 = row[base + 1] * rs * nw[base + 1];
    int pidx = (c < 22) ? t_idx : ((c < 43) ? h_idx : w_idx);
    float ang = freqs[pidx * C2 + c];
    float cs = __cosf(ang), sn = __sinf(ang);
    outb[(size_t)s * DIM + base]     = (__bf16)(x0 * cs - x1 * sn);
    outb[(size_t)s * DIM + base + 1] = (__bf16)(x0 * sn + x1 * cs);
  }
}

// ---------------------------------------------------------------- flash attn
// Block = (q-tile of 32 rows, one head). 128 threads / 4 waves.
// Score phase: wave(sm,sn) computes one 16x16 frag of the 32x32 score tile.
// PV phase: wave owns 32 output columns (dc0 = wave*32), 2x2 f32 frags.
// Q tile + double-buffered K tiles are DMA'd by the TDM (hardware zero-fill
// on the causal tail via tensor_dim1 = jmax - j0); V needs a transpose ->
// staged manually. s_wait_tensorcnt 1 tolerates the in-flight next K tile.
__global__ __launch_bounds__(128) void flash_attn_kernel(
    const __bf16* __restrict__ Qg, const __bf16* __restrict__ Kg,
    const __bf16* __restrict__ Vg, float* __restrict__ Og) {
  __shared__ __bf16 Qs[32][128];
  __shared__ __bf16 Ks[2][32][128];
  __shared__ __bf16 Vt[128][32];     // transposed: [d][j]
  __shared__ float  Ps[32][33];
  __shared__ __bf16 Pb[32][32];
  __shared__ float  mrow[32], lrow[32], arow[32];

  const int tid = threadIdx.x, lane = tid & 31, wave = tid >> 5;
  const int sm = wave & 1, sn = wave >> 1;
  const int dc0 = wave * 32;
  const int q0 = blockIdx.x * 32;
  const int h  = blockIdx.y;
  const float scale = 0.088388347648318447f;  // 1/sqrt(128)

  int qlast = q0 + 31; if (qlast > S - 1) qlast = S - 1;
  const int jmax = (qlast / HW + 1) * HW;       // causal frame bound
  const int nkt = (jmax + 31) >> 5;

  auto issue_k = [&](int jt, int p) {
    const int j0 = jt << 5;
    tdm_load_2d_b16(lds_addr_of(&Ks[p][0][0]), Kg + (size_t)j0 * DIM + h * HD,
                    (unsigned)(DIM - h * HD), (unsigned)(jmax - j0),
                    (unsigned)DIM, 128u, 32u, false);
  };

  // stage Q tile (32 x 128) via TDM; rows beyond S zero-filled by hardware,
  // then prefetch the first K tile.
  if (wave == 0) {
    tdm_load_2d_b16(lds_addr_of(&Qs[0][0]), Qg + (size_t)q0 * DIM + h * HD,
                    (unsigned)(DIM - h * HD), (unsigned)(S - q0), (unsigned)DIM,
                    128u, 32u, false);
    issue_k(0, 0);
  }
  if (tid < 32) { mrow[tid] = -3.0e38f; lrow[tid] = 0.f; }

  v8f occ[2][2];
#pragma unroll
  for (int i = 0; i < 2; ++i)
#pragma unroll
    for (int j = 0; j < 2; ++j) occ[i][j] = v8f_zero();

  for (int jt = 0; jt < nkt; ++jt) {
    const int p = jt & 1;
    const int j0 = jt << 5;
    if (wave == 0) {
      if (jt + 1 < nkt) {
        issue_k(jt + 1, p ^ 1);                        // prefetch next K tile
        __builtin_amdgcn_s_wait_tensorcnt((short)1);   // Q + K(jt) complete
      } else {
        __builtin_amdgcn_s_wait_tensorcnt((short)0);
      }
    }
    // stage V transposed [d][j] (B layout for PV: n=d, k=j)
    for (int i = tid; i < 4096; i += 128) {
      int j = i >> 7, d = i & 127;
      __bf16 val = (__bf16)0.f;
      if (j0 + j < jmax) val = Vg[(size_t)(j0 + j) * DIM + h * HD + d];
      Vt[d][j] = val;
    }
    __syncthreads();

    // scores: 16x16 frag per wave, K-dim = 128 in 4 steps
    v8f sc = v8f_zero();
#pragma unroll
    for (int kc = 0; kc < 4; ++kc) {
      v16bf a = load_frag_a(&Qs[sm * 16][kc * 32], 128);
      v16bf b = load_frag_b(&Ks[p][sn * 16][kc * 32], 128);
      sc = wmma_bf16(a, b, sc);
    }
    {
      int n  = sn * 16 + (lane & 15);
      int jf = (j0 + n) / HW;
#pragma unroll
      for (int v = 0; v < 8; ++v) {
        int m  = sm * 16 + v + ((lane >> 4) << 3);
        int qf = (q0 + m) / HW;
        float val = sc[v] * scale;
        if (jf > qf) val = -3.0e38f;
        Ps[m][n] = val;
      }
    }
    __syncthreads();

    // streaming softmax row pass (one lane per q row)
    if (tid < 32) {
      int r = tid;
      float rm = Ps[r][0];
      for (int j = 1; j < 32; ++j) rm = fmaxf(rm, Ps[r][j]);
      float mold = mrow[r];
      float mnew = fmaxf(mold, rm);
      float alpha = __expf(mold - mnew);
      float sum = 0.f;
      for (int j = 0; j < 32; ++j) {
        float p2 = __expf(Ps[r][j] - mnew);
        Pb[r][j] = (__bf16)p2;
        sum += p2;
      }
      lrow[r] = lrow[r] * alpha + sum;
      arow[r] = alpha;
      mrow[r] = mnew;
    }
    __syncthreads();

    // rescale running O, then PV
#pragma unroll
    for (int mi = 0; mi < 2; ++mi)
#pragma unroll
      for (int ni = 0; ni < 2; ++ni)
#pragma unroll
        for (int v = 0; v < 8; ++v) {
          int m = mi * 16 + v + ((lane >> 4) << 3);
          occ[mi][ni][v] *= arow[m];
        }
    v16bf aP0 = load_frag_a(&Pb[0][0],  32);
    v16bf aP1 = load_frag_a(&Pb[16][0], 32);
#pragma unroll
    for (int ni = 0; ni < 2; ++ni) {
      v16bf bV = load_frag_b(&Vt[dc0 + ni * 16][0], 32);
      occ[0][ni] = wmma_bf16(aP0, bV, occ[0][ni]);
      occ[1][ni] = wmma_bf16(aP1, bV, occ[1][ni]);
    }
    __syncthreads();   // Vt + Ks[p] reads done before next iteration refills
  }

  // epilogue: normalize and store
#pragma unroll
  for (int mi = 0; mi < 2; ++mi)
#pragma unroll
    for (int ni = 0; ni < 2; ++ni)
#pragma unroll
      for (int v = 0; v < 8; ++v) {
        int m = mi * 16 + v + ((lane >> 4) << 3);
        int qs = q0 + m;
        if (qs < S) {
          int d = dc0 + ni * 16 + (lane & 15);
          Og[(size_t)qs * DIM + h * HD + d] = occ[mi][ni][v] / lrow[m];
        }
      }
}

// ---------------------------------------------------------------- launch
extern "C" void kernel_launch(void* const* d_in, const int* in_sizes, int n_in,
                              void* d_out, int out_size, void* d_ws, size_t ws_size,
                              hipStream_t stream) {
  (void)in_sizes; (void)n_in; (void)out_size; (void)ws_size;
  const float* x        = (const float*)d_in[0];
  const float* q_w      = (const float*)d_in[1];
  const float* q_b      = (const float*)d_in[2];
  const float* k_w      = (const float*)d_in[3];
  const float* k_b      = (const float*)d_in[4];
  const float* v_w      = (const float*)d_in[5];
  const float* v_b      = (const float*)d_in[6];
  const float* o_w      = (const float*)d_in[7];
  const float* o_b      = (const float*)d_in[8];
  const float* norm_q_w = (const float*)d_in[9];
  const float* norm_k_w = (const float*)d_in[10];
  const float* freqs    = (const float*)d_in[11];
  float* out = (float*)d_out;

  char* ws = (char*)d_ws;
  size_t off = 0;
  auto alloc = [&](size_t bytes) -> void* {
    void* p = ws + off;
    off = (off + bytes + 255) & ~(size_t)255;
    return p;
  };
  __bf16* xb   = (__bf16*)alloc((size_t)S * DIM * 2);
  __bf16* qwb  = (__bf16*)alloc((size_t)DIM * DIM * 2);
  __bf16* kwb  = (__bf16*)alloc((size_t)DIM * DIM * 2);
  __bf16* vwb  = (__bf16*)alloc((size_t)DIM * DIM * 2);
  __bf16* owb  = (__bf16*)alloc((size_t)DIM * DIM * 2);
  float*  raw  = (float*)alloc((size_t)S * DIM * 4);   // reused for Q/K/V raw
  __bf16* Qb   = (__bf16*)alloc((size_t)S * DIM * 2);
  __bf16* Kb   = (__bf16*)alloc((size_t)S * DIM * 2);
  __bf16* Vb   = (__bf16*)alloc((size_t)S * DIM * 2);
  float*  Obuf = (float*)alloc((size_t)S * DIM * 4);
  __bf16* Ob   = (__bf16*)alloc((size_t)S * DIM * 2);

  const int nX = S * DIM;
  const int nW = DIM * DIM;
  f32_to_bf16_kernel<<<256, 256, 0, stream>>>(x,   xb,  nX);
  f32_to_bf16_kernel<<<512, 256, 0, stream>>>(q_w, qwb, nW);
  f32_to_bf16_kernel<<<512, 256, 0, stream>>>(k_w, kwb, nW);
  f32_to_bf16_kernel<<<512, 256, 0, stream>>>(v_w, vwb, nW);
  f32_to_bf16_kernel<<<512, 256, 0, stream>>>(o_w, owb, nW);

  dim3 ggrid((S + 63) / 64, DIM / 64);

  // Q path
  gemm_bf16_wmma<<<ggrid, 128, 0, stream>>>(xb, qwb, q_b, raw, S, DIM, DIM);
  rmsnorm_rope_kernel<<<S, 256, 0, stream>>>(raw, norm_q_w, freqs, Qb);
  // K path
  gemm_bf16_wmma<<<ggrid, 128, 0, stream>>>(xb, kwb, k_b, raw, S, DIM, DIM);
  rmsnorm_rope_kernel<<<S, 256, 0, stream>>>(raw, norm_k_w, freqs, Kb);
  // V path
  gemm_bf16_wmma<<<ggrid, 128, 0, stream>>>(xb, vwb, v_b, raw, S, DIM, DIM);
  f32_to_bf16_kernel<<<256, 256, 0, stream>>>(raw, Vb, nX);

  // attention
  dim3 agrid((S + 31) / 32, NH);
  flash_attn_kernel<<<agrid, 128, 0, stream>>>(Qb, Kb, Vb, Obuf);

  // output projection
  f32_to_bf16_kernel<<<256, 256, 0, stream>>>(Obuf, Ob, nX);
  gemm_bf16_wmma<<<ggrid, 128, 0, stream>>>(Ob, owb, o_b, out, S, DIM, DIM);
}
